// SingleHeadAttention_55997783605341
// MI455X (gfx1250) — matrix operand order repaired and checked by reading, not compiled, not run on hardware
//
#include <hip/hip_runtime.h>
#include <hip/hip_bf16.h>

typedef _Float16 v16h __attribute__((ext_vector_type(16)));
typedef _Float16 v8h  __attribute__((ext_vector_type(8)));
typedef float    v8f  __attribute__((ext_vector_type(8)));

namespace {
constexpr int kB   = 2;
constexpr int kS   = 2048;
constexpr int kHid = 2048;
constexpr int kNH  = 32;
constexpr int kNKV = 8;
constexpr int kHD  = 64;
constexpr int kM   = kB * kS;        // 4096 token rows
constexpr int kKVD = kNKV * kHD;     // 512
}

__device__ __forceinline__ v16h cat16(v8h lo, v8h hi) {
  v16h r;
#pragma unroll
  for (int i = 0; i < 8; ++i) { r[i] = lo[i]; r[i + 8] = hi[i]; }
  return r;
}

__device__ __forceinline__ v8f wmma16(v16h a, v16h b, v8f c) {
  // D = A(16x32 f16) * B(32x16 f16) + C(16x16 f32)
  return __builtin_amdgcn_wmma_f32_16x16x32_f16(false, a, false, b, (short)0, c,
                                                false, false);
}

// ---------------------------------------------------------------- elementwise
__global__ void cvt_f32_f16(const float* __restrict__ in,
                            _Float16* __restrict__ out, int n) {
  int i = blockIdx.x * blockDim.x + threadIdx.x;
  if (i < n) out[i] = (_Float16)in[i];
}

// In-place RoPE on projected tensor laid out [B, S, nheads, HD].
// `scale` lets us fold the attention 1/sqrt(HD) into Q for free.
__global__ void rope_f16(_Float16* __restrict__ x,
                         const float* __restrict__ rope_real,
                         const float* __restrict__ rope_im, int nheads,
                         float scale) {
  int idx = blockIdx.x * blockDim.x + threadIdx.x;
  int total = kB * kS * nheads * (kHD / 2);
  if (idx >= total) return;
  int d  = idx % (kHD / 2);
  int t  = idx / (kHD / 2);          // (b*S + s)*nheads + h
  int t2 = t / nheads;
  int s  = t2 % kS;
  size_t base = (size_t)t * kHD;
  float xr = (float)x[base + d];
  float xi = (float)x[base + 32 + d];
  float c  = rope_real[s * 32 + d];
  float sn = rope_im[s * 32 + d];
  x[base + d]      = (_Float16)((xr * c - xi * sn) * scale);
  x[base + 32 + d] = (_Float16)((xr * sn + xi * c) * scale);
}

// V: [B, S, NKV, HD] -> Vt: [B, NKV, HD, S]  (so flash attention reads rows)
__global__ void vtrans_f16(const _Float16* __restrict__ v,
                           _Float16* __restrict__ vt) {
  int idx = blockIdx.x * blockDim.x + threadIdx.x;
  int total = kB * kNKV * kHD * kS;
  if (idx >= total) return;
  int s  = idx % kS;
  int t  = idx / kS;
  int d  = t % kHD;
  int t2 = t / kHD;
  int h  = t2 % kNKV;
  int b  = t2 / kNKV;
  vt[idx] = v[(((size_t)b * kS + s) * kNKV + h) * kHD + d];
}

// ---------------------------------------------------------------- WMMA GEMM
// Y[M,N] = X[M,K] * W[N,K]^T.  One wave computes a 64(M) x 64(N) tile:
// per 32-deep k-step it loads 4 A-fragments + 4 B-fragments and issues 16
// WMMAs, giving ~4x the arithmetic intensity of a 16x64 strip.
template <bool F32OUT>
__global__ __launch_bounds__(128) void gemm_xwT(
    const _Float16* __restrict__ X, const _Float16* __restrict__ W,
    void* __restrict__ Yv, int M, int N, int K) {
  const int lane = threadIdx.x & 31;
  const int wave = threadIdx.x >> 5;
  const int hg   = lane >> 4;        // half-group 0/1
  const int l15  = lane & 15;
  const int m0 = blockIdx.y * 64;
  const int n0 = blockIdx.x * 256 + wave * 64;

  v8f acc[4][4] = {};
  const _Float16* xbase = X + (size_t)(m0 + l15) * K;
  const _Float16* wbase = W + (size_t)(n0 + l15) * K;
  for (int k0 = 0; k0 < K; k0 += 32) {
    v16h bf[4];
#pragma unroll
    for (int t = 0; t < 4; ++t)  // B: lane = col n, contiguous 16 f16 along K
      bf[t] = *(const v16h*)(wbase + (size_t)t * 16 * K + k0 + hg * 16);
#pragma unroll
    for (int mm = 0; mm < 4; ++mm) {
      const _Float16* ar = xbase + (size_t)mm * 16 * K + k0;
      v16h a = cat16(*(const v8h*)(ar + hg * 16),
                     *(const v8h*)(ar + hg * 16 + 8));
#pragma unroll
      for (int t = 0; t < 4; ++t) acc[mm][t] = wmma16(a, bf[t], acc[mm][t]);
    }
  }
  // C layout: VGPR v holds row (mm*16 + v + hg*8), col n = l15
#pragma unroll
  for (int mm = 0; mm < 4; ++mm)
#pragma unroll
    for (int t = 0; t < 4; ++t) {
      int n = n0 + t * 16 + l15;
#pragma unroll
      for (int v = 0; v < 8; ++v) {
        int m = m0 + mm * 16 + hg * 8 + v;
        if (F32OUT)
          ((float*)Yv)[(size_t)m * N + n] = acc[mm][t][v];
        else
          ((_Float16*)Yv)[(size_t)m * N + n] = (_Float16)acc[mm][t][v];
      }
    }
}

// ---------------------------------------------------------------- attention
// Flash attention computed transposed: scores^T = K*Q^T, O^T = V^T*P^T.
// One wave owns (b, qh, 16 query rows); lane owns one query column, so the
// online-softmax stats are per-lane scalars.  Q is pre-scaled by 1/sqrt(HD).
// The key loop is split into (q0>>5) fully-unmasked blocks plus exactly one
// masked diagonal block, so the hot loop carries no compares/selects.
__global__ __launch_bounds__(256) void attn_fwd(
    const _Float16* __restrict__ Q,   // [B,S,NH,HD] (RoPE'd, pre-scaled)
    const _Float16* __restrict__ Km,  // [B,S,NKV,HD] (RoPE'd)
    const _Float16* __restrict__ Vt,  // [B,NKV,HD,S]
    _Float16* __restrict__ O) {       // [B,S,NH,HD]
  const int lane = threadIdx.x & 31;
  const int hg   = lane >> 4;
  const int l15  = lane & 15;
  const int wid  = blockIdx.x * (blockDim.x >> 5) + (threadIdx.x >> 5);
  const int qt  = wid % (kS / 16);
  int tq        = wid / (kS / 16);
  const int qh  = tq % kNH;
  const int b   = tq / kNH;
  const int kvh = qh >> 2;           // NH/NKV = 4 (repeat_interleave)
  const int q0  = qt * 16;
  const int qrow = q0 + l15;

  // Q B-fragments (contraction dim = head dim, chunks 0..31 / 32..63)
  const _Float16* qptr = Q + ((size_t)b * kS + qrow) * (kNH * kHD) + qh * kHD;
  v16h bq0 = cat16(*(const v8h*)(qptr + hg * 8), *(const v8h*)(qptr + 16 + hg * 8));
  v16h bq1 = cat16(*(const v8h*)(qptr + 32 + hg * 8), *(const v8h*)(qptr + 48 + hg * 8));

  const _Float16* kbase = Km + (size_t)b * kS * kKVD + kvh * kHD;
  const _Float16* vbase = Vt + ((size_t)(b * kNKV + kvh) * kHD + l15) * kS;

  v8f  oacc[4] = {};                 // O^T: 4 hd-tiles of 16x16
  float mi = -3.0e38f, li = 0.0f;

  auto do_block = [&](int kb, bool masked) {
    // scores^T tiles: rows = keys, cols = queries
    v8f st0 = {}, st1 = {};
#pragma unroll
    for (int tt = 0; tt < 2; ++tt) {
      const _Float16* kptr = kbase + (size_t)(kb + tt * 16 + l15) * kKVD;
      v16h ak0 = cat16(*(const v8h*)(kptr + hg * 8),
                       *(const v8h*)(kptr + 16 + hg * 8));
      v16h ak1 = cat16(*(const v8h*)(kptr + 32 + hg * 8),
                       *(const v8h*)(kptr + 48 + hg * 8));
      if (tt == 0) { st0 = wmma16(ak0, bq0, st0); st0 = wmma16(ak1, bq1, st0); }
      else         { st1 = wmma16(ak0, bq0, st1); st1 = wmma16(ak1, bq1, st1); }
    }

    // online softmax (per-lane = per-query column); mask only when needed
    float x0[8], x1[8];
    float mloc = -3.0e38f;
#pragma unroll
    for (int v = 0; v < 8; ++v) {
      float a0 = st0[v], a1 = st1[v];        // C row m = v + hg*8
      if (masked) {
        int key0 = kb + v + hg * 8;
        if (key0 > qrow)      a0 = -1.0e9f;
        if (key0 + 16 > qrow) a1 = -1.0e9f;
      }
      x0[v] = a0; x1[v] = a1;
      mloc = fmaxf(mloc, fmaxf(a0, a1));
    }
    mloc = fmaxf(mloc, __shfl_xor(mloc, 16));
    float mnew  = fmaxf(mi, mloc);
    float alpha = __expf(mi - mnew);
    float ssum = 0.0f;
#pragma unroll
    for (int v = 0; v < 8; ++v) {
      x0[v] = __expf(x0[v] - mnew);
      x1[v] = __expf(x1[v] - mnew);
      ssum += x0[v] + x1[v];
    }
    ssum += __shfl_xor(ssum, 16);
    li = li * alpha + ssum;
    mi = mnew;
#pragma unroll
    for (int h = 0; h < 4; ++h)
#pragma unroll
      for (int v = 0; v < 8; ++v) oacc[h][v] *= alpha;

    // P^T B-fragment: lane = query col; needs keys hg*16..hg*16+15.
    // Own half supplies 8 of them; the other 8 come via one xor-16 exchange.
    v16h bp;
#pragma unroll
    for (int v = 0; v < 8; ++v) {
      float s0 = __shfl_xor(x0[v], 16);
      float s1 = __shfl_xor(x1[v], 16);
      if (hg == 0) { bp[v] = (_Float16)x0[v]; bp[8 + v] = (_Float16)s0; }
      else         { bp[v] = (_Float16)s1;    bp[8 + v] = (_Float16)x1[v]; }
    }

    // O^T += V^T * P^T, 4 hd-tiles; V^T rows are contiguous in Vt
#pragma unroll
    for (int h = 0; h < 4; ++h) {
      const _Float16* vptr = vbase + (size_t)(h * 16) * kS + kb;
      v16h av = cat16(*(const v8h*)(vptr + hg * 16),
                      *(const v8h*)(vptr + hg * 16 + 8));
      oacc[h] = wmma16(av, bp, oacc[h]);
    }
  };

  const int nfull = q0 >> 5;              // fully-unmasked 32-key blocks
  for (int i = 0; i < nfull; ++i) do_block(i * 32, false);
  do_block(nfull * 32, true);             // single masked diagonal block

  // normalize (per-lane = per-query) and store O^T back row-major
  float rcp = 1.0f / li;
  _Float16* optr = O + ((size_t)b * kS + qrow) * (kNH * kHD) + qh * kHD;
#pragma unroll
  for (int h = 0; h < 4; ++h)
#pragma unroll
    for (int v = 0; v < 8; ++v)
      optr[h * 16 + hg * 8 + v] = (_Float16)(oacc[h][v] * rcp);
}

// ---------------------------------------------------------------- launcher
extern "C" void kernel_launch(void* const* d_in, const int* in_sizes, int n_in,
                              void* d_out, int out_size, void* d_ws,
                              size_t ws_size, hipStream_t stream) {
  (void)in_sizes; (void)n_in; (void)out_size; (void)ws_size;
  const float* hidden    = (const float*)d_in[0];
  /* d_in[1] = attention_mask (causal; computed analytically) */
  const float* rope_real = (const float*)d_in[2];
  const float* rope_im   = (const float*)d_in[3];
  const float* Wq        = (const float*)d_in[4];
  const float* Wk        = (const float*)d_in[5];
  const float* Wv        = (const float*)d_in[6];
  const float* Wo        = (const float*)d_in[7];
  float* out = (float*)d_out;

  char* p = (char*)d_ws;
  _Float16* Xh  = (_Float16*)p; p += (size_t)kM * kHid * 2;
  _Float16* Wqh = (_Float16*)p; p += (size_t)kHid * kHid * 2;
  _Float16* Wkh = (_Float16*)p; p += (size_t)kKVD * kHid * 2;
  _Float16* Wvh = (_Float16*)p; p += (size_t)kKVD * kHid * 2;
  _Float16* Woh = (_Float16*)p; p += (size_t)kHid * kHid * 2;
  _Float16* Qt  = (_Float16*)p; p += (size_t)kM * kHid * 2;   // [B,S,NH,HD]
  _Float16* Kt  = (_Float16*)p; p += (size_t)kM * kKVD * 2;   // [B,S,NKV,HD]
  _Float16* Vm  = (_Float16*)p; p += (size_t)kM * kKVD * 2;   // [B,S,NKV,HD]
  _Float16* Vtr = (_Float16*)p; p += (size_t)kM * kKVD * 2;   // [B,NKV,HD,S]
  _Float16* Ao  = (_Float16*)p; p += (size_t)kM * kHid * 2;   // [B,S,NH*HD]

  auto cvt = [&](const float* src, _Float16* dst, int n) {
    cvt_f32_f16<<<(n + 255) / 256, 256, 0, stream>>>(src, dst, n);
  };
  cvt(hidden, Xh, kM * kHid);
  cvt(Wq, Wqh, kHid * kHid);
  cvt(Wk, Wkh, kKVD * kHid);
  cvt(Wv, Wvh, kKVD * kHid);
  cvt(Wo, Woh, kHid * kHid);

  // projections: one wave per 64x64 tile, 4 waves (64x256) per block
  gemm_xwT<false><<<dim3(kHid / 256, kM / 64), 128, 0, stream>>>(
      Xh, Wqh, Qt, kM, kHid, kHid);
  gemm_xwT<false><<<dim3(kKVD / 256, kM / 64), 128, 0, stream>>>(
      Xh, Wkh, Kt, kM, kKVD, kHid);
  gemm_xwT<false><<<dim3(kKVD / 256, kM / 64), 128, 0, stream>>>(
      Xh, Wvh, Vm, kM, kKVD, kHid);

  // RoPE; fold the attention scale 1/sqrt(64) into Q
  int nq = kB * kS * kNH * (kHD / 2);
  rope_f16<<<(nq + 255) / 256, 256, 0, stream>>>(Qt, rope_real, rope_im, kNH,
                                                 0.125f);
  int nk = kB * kS * kNKV * (kHD / 2);
  rope_f16<<<(nk + 255) / 256, 256, 0, stream>>>(Kt, rope_real, rope_im, kNKV,
                                                 1.0f);

  int nv = kB * kNKV * kHD * kS;
  vtrans_f16<<<(nv + 255) / 256, 256, 0, stream>>>(Vm, Vtr);

  // flash attention: 8192 waves = 1024 blocks of 8 waves
  int nwaves = kB * kNH * (kS / 16);
  attn_fwd<<<nwaves / 8, 256, 0, stream>>>(Qt, Kt, Vtr, Ao);

  // output projection, fp32 result straight to d_out
  gemm_xwT<true><<<dim3(kHid / 256, kM / 64), 128, 0, stream>>>(
      Ao, Woh, out, kM, kHid, kHid);
}
